// InteractionBlock_42004780155369
// MI455X (gfx1250) — matrix operand order repaired and checked by reading, not compiled, not run on hardware
//
#include <hip/hip_runtime.h>
#include <cstdint>
#include <cstddef>

// ---------------------------------------------------------------------------
// Problem constants (match reference)
// ---------------------------------------------------------------------------
#define E_EDGES   262144
#define T_TRIP    2097152
#define EMBED     128
#define ANGLE     64
#define BASIS     8
#define RBF_DIM   6
#define SBF_DIM   42

typedef __bf16 bf16_t;
typedef __attribute__((ext_vector_type(16))) __bf16 v16bf;
typedef __attribute__((ext_vector_type(8)))  float  v8f;
typedef __attribute__((ext_vector_type(4)))  int    v4i_t;

__device__ __forceinline__ float swishf(float x) {
    return x / (1.0f + __expf(-x));
}

// ---------------------------------------------------------------------------
// Generic fused WMMA GEMM:  out = epi( A[M,K](bf16) @ W[K,N] + bias )
// Weights are stored pre-transposed, N-major: WT[n*K + k] (bf16).
// EPI: 0 = none, 1 = swish, 2 = swish then *aux, 3 = swish then +aux
// A tile is staged into LDS via GLOBAL_LOAD_ASYNC_TO_LDS (ASYNCcnt) when the
// toolchain declares the builtin; otherwise via VGPR-staged b128 copies.
// ---------------------------------------------------------------------------
template<int K, int N, bool BIAS, int EPI, bool OUTF32>
__global__ __launch_bounds__(256) void gemm_bf16_k(
    const bf16_t* __restrict__ A,
    const bf16_t* __restrict__ WT,
    const float*  __restrict__ bias,
    const bf16_t* __restrict__ aux,
    void*         __restrict__ outp)
{
    __shared__ bf16_t As[128 * K];           // 32KB (K=128) or 16KB (K=64)

    const int tid  = threadIdx.x;
    const int wave = tid >> 5;               // wave32: 8 waves / block
    const int lane = tid & 31;
    const long long blockRow = (long long)blockIdx.x * 128;

    // Stage the A tile (128 x K bf16) into LDS.
    {
        const int nq = 128 * K / 8;          // 8 bf16 per 16-byte chunk
#if defined(__gfx1250__) && __has_builtin(__builtin_amdgcn_global_load_async_to_lds_b128)
        // CDNA5 async copy: memory -> LDS directly, tracked on ASYNCcnt,
        // no VGPR round-trip, co-executes with surrounding VALU/WMMA.
        for (int i = tid; i < nq; i += 256) {
            v4i_t* gsrc = (v4i_t*)(A + blockRow * K + (long long)i * 8);
            v4i_t* ldst = (v4i_t*)(As + i * 8);
            __builtin_amdgcn_global_load_async_to_lds_b128(gsrc, ldst, 0, 0);
        }
#if __has_builtin(__builtin_amdgcn_s_wait_asynccnt)
        __builtin_amdgcn_s_wait_asynccnt(0);
#else
        asm volatile("s_wait_asynccnt 0x0" ::: "memory");
#endif
#else
        const uint4* src = (const uint4*)(A + blockRow * K);
        uint4*       dst = (uint4*)As;
        for (int i = tid; i < nq; i += 256) dst[i] = src[i];
#endif
    }
    __syncthreads();

    constexpr int NT = N / 16;               // N tiles per wave strip
    constexpr int KS = K / 32;               // K steps of 32

    const int mloc = wave * 16 + (lane & 15);    // local row this lane feeds
    const int half = (lane < 16) ? 0 : 1;

    // Prefetch this lane's epilogue aux row while the WMMA chain runs.
    if (EPI >= 2) {
        const long long m0 = blockRow + wave * 16 + half * 8;
        __builtin_prefetch(aux + m0 * N, 0, 3);
    }

    v8f acc[NT];
#pragma unroll
    for (int nt = 0; nt < NT; ++nt) {
        v8f z = {0.f, 0.f, 0.f, 0.f, 0.f, 0.f, 0.f, 0.f};
        acc[nt] = z;
    }

#pragma unroll
    for (int ks = 0; ks < KS; ++ks) {
        // A fragment: ISA 16-bit A 16x32 layout.
        // lane<16 : K = ks*32 + {0..7, 16..23}; lane>=16 : K = ks*32 + {8..15, 24..31}
        union { uint4 q[2]; v16bf v; } af;
        const int ka = ks * 32 + half * 8;
        af.q[0] = *(const uint4*)(As + mloc * K + ka);
        af.q[1] = *(const uint4*)(As + mloc * K + ka + 16);

        const int kb = ks * 32 + half * 8;       // B: lane<16 K {0..7,16..23}, lane>=16 K {8..15,24..31}
#pragma unroll
        for (int nt = 0; nt < NT; ++nt) {
            union { uint4 q[2]; v16bf v; } bfg;
            const int n = nt * 16 + (lane & 15); // lane selects column
            const uint4* wp = (const uint4*)(WT + (long long)n * K + kb);
            bfg.q[0] = wp[0];
            bfg.q[1] = wp[2];
            acc[nt] = __builtin_amdgcn_wmma_f32_16x16x32_bf16(
                false, af.v, false, bfg.v, (short)0, acc[nt], false, false);
        }
    }

    // Epilogue: C/D layout -> VGPR i: lane<16 M=i, lane>=16 M=8+i; N=lane&15.
#pragma unroll
    for (int nt = 0; nt < NT; ++nt) {
        const int n = nt * 16 + (lane & 15);
        const float bv = BIAS ? bias[n] : 0.0f;
#pragma unroll
        for (int i = 0; i < 8; ++i) {
            const long long m = blockRow + wave * 16 + half * 8 + i;
            float x = acc[nt][i] + bv;
            if (EPI >= 1) x = swishf(x);
            if (EPI == 2) x *= (float)aux[m * N + n];
            if (EPI == 3) x += (float)aux[m * N + n];
            if (OUTF32) ((float*)outp)[m * N + n] = x;
            else        ((bf16_t*)outp)[m * N + n] = (bf16_t)x;
        }
    }
}

// ---------------------------------------------------------------------------
// Weight prep kernels
// ---------------------------------------------------------------------------
// Transpose f32 W[K,N] -> bf16 WT[n*K+k] (B-fragment friendly layout).
__global__ void transpose_to_bf16_k(const float* __restrict__ W,
                                    bf16_t* __restrict__ WT, int K, int N)
{
    int idx = blockIdx.x * 256 + threadIdx.x;
    if (idx < K * N) {
        int k = idx / N, n = idx % N;
        WT[n * K + k] = (bf16_t)W[idx];
    }
}

// Wc[6,128] = W_rbf1[6,8] @ W_rbf2[8,128]  (f32, tiny)
__global__ void rbf_combine_k(const float* __restrict__ W1,
                              const float* __restrict__ W2,
                              float* __restrict__ Wc)
{
    int idx = blockIdx.x * 256 + threadIdx.x;
    if (idx < RBF_DIM * EMBED) {
        int k = idx / EMBED, n = idx % EMBED;
        float s = 0.f;
#pragma unroll
        for (int j = 0; j < BASIS; ++j) s += W1[k * BASIS + j] * W2[j * EMBED + n];
        Wc[idx] = s;
    }
}

// ---------------------------------------------------------------------------
// Elementwise / conversion kernels
// ---------------------------------------------------------------------------
__global__ void f32_to_bf16_k(const float* __restrict__ in,
                              bf16_t* __restrict__ out, long long n)
{
    long long i = (long long)blockIdx.x * blockDim.x + threadIdx.x;
    long long stride = (long long)gridDim.x * blockDim.x;
    for (; i < n; i += stride) out[i] = (bf16_t)in[i];
}

__global__ void f64_to_bf16_k(const double* __restrict__ in,
                              bf16_t* __restrict__ out, long long n)
{
    long long i = (long long)blockIdx.x * blockDim.x + threadIdx.x;
    long long stride = (long long)gridDim.x * blockDim.x;
    for (; i < n; i += stride) out[i] = (bf16_t)(float)in[i];
}

__global__ void zero_f64_k(double* __restrict__ p, long long n)
{
    long long i = (long long)blockIdx.x * blockDim.x + threadIdx.x;
    long long stride = (long long)gridDim.x * blockDim.x;
    for (; i < n; i += stride) p[i] = 0.0;
}

// rbf_e[E,128] = rbf[E,6] @ Wc[6,128], stored bf16 (epilogue scale operand)
__global__ void rbf_expand_k(const float* __restrict__ rbf,
                             const float* __restrict__ Wc,
                             bf16_t* __restrict__ out)
{
    __shared__ float Ws[RBF_DIM * EMBED];
    for (int i = threadIdx.x; i < RBF_DIM * EMBED; i += blockDim.x) Ws[i] = Wc[i];
    __syncthreads();
    const int n = threadIdx.x & 127;
    const long long e = (long long)blockIdx.x * 2 + (threadIdx.x >> 7);
    float r[RBF_DIM];
#pragma unroll
    for (int k = 0; k < RBF_DIM; ++k) r[k] = rbf[e * RBF_DIM + k];
    float s = 0.f;
#pragma unroll
    for (int k = 0; k < RBF_DIM; ++k) s += r[k] * Ws[k * EMBED + n];
    out[e * EMBED + n] = (bf16_t)s;
}

// ---------------------------------------------------------------------------
// Triplet stage: sbf -> basis -> angle scale, gather m_down, f64 scatter-sum.
// One thread per triplet; m_down table (32MB bf16) stays L2-resident;
// agg accumulator (128MB f64) also L2-resident -> fast f64 atomics.
// ---------------------------------------------------------------------------
__global__ __launch_bounds__(256) void triplet_k(
    const float* __restrict__ sbf,
    const int*   __restrict__ reduce_to_ji,
    const int*   __restrict__ expand_to_kj,
    const bf16_t* __restrict__ m_down,
    const float* __restrict__ Wsbf1,   // [42,8]
    const float* __restrict__ Wsbf2,   // [8,64]
    double*      __restrict__ agg)     // [E,64]
{
    __shared__ float w1[SBF_DIM * BASIS];
    __shared__ float w2[BASIS * ANGLE];
    for (int i = threadIdx.x; i < SBF_DIM * BASIS; i += blockDim.x) w1[i] = Wsbf1[i];
    for (int i = threadIdx.x; i < BASIS * ANGLE; i += blockDim.x)  w2[i] = Wsbf2[i];
    __syncthreads();

    const long long t = (long long)blockIdx.x * blockDim.x + threadIdx.x;
    if (t >= T_TRIP) return;

    const float* s = sbf + t * SBF_DIM;
    float h[BASIS];
#pragma unroll
    for (int j = 0; j < BASIS; ++j) h[j] = 0.f;
    for (int k = 0; k < SBF_DIM; ++k) {
        const float sv = s[k];
#pragma unroll
        for (int j = 0; j < BASIS; ++j) h[j] += sv * w1[k * BASIS + j];
    }

    const long long esrc = expand_to_kj[t];
    const long long edst = reduce_to_ji[t];
    const bf16_t* md = m_down + esrc * ANGLE;
    double* ag = agg + edst * ANGLE;

#pragma unroll 4
    for (int n = 0; n < ANGLE; ++n) {
        float e = 0.f;
#pragma unroll
        for (int j = 0; j < BASIS; ++j) e += h[j] * w2[j * ANGLE + n];
        const float v = (float)md[n] * e;
        atomicAdd(ag + n, (double)v);       // GLOBAL_ATOMIC_ADD_F64
    }
}

// ---------------------------------------------------------------------------
// Launch
// ---------------------------------------------------------------------------
extern "C" void kernel_launch(void* const* d_in, const int* in_sizes, int n_in,
                              void* d_out, int out_size, void* d_ws, size_t ws_size,
                              hipStream_t stream)
{
    (void)in_sizes; (void)n_in; (void)out_size; (void)ws_size;

    // Inputs in setup_inputs() order (depth-first through params)
    const float* m_input  = (const float*)d_in[0];
    const float* rbf      = (const float*)d_in[1];
    const float* sbf      = (const float*)d_in[2];
    const int*   red_ji   = (const int*)  d_in[3];
    const int*   exp_kj   = (const int*)  d_in[4];
    const float* W_rbf1   = (const float*)d_in[5];
    const float* W_rbf2   = (const float*)d_in[6];
    const float* W_sbf1   = (const float*)d_in[7];
    const float* W_sbf2   = (const float*)d_in[8];
    const float* W_kj     = (const float*)d_in[9];
    const float* b_kj     = (const float*)d_in[10];
    const float* W_down   = (const float*)d_in[11];
    const float* W_up     = (const float*)d_in[12];
    const float* W_ji     = (const float*)d_in[13];
    const float* b_ji     = (const float*)d_in[14];
    const float* rbW1     = (const float*)d_in[15];
    const float* rb_b1    = (const float*)d_in[16];
    const float* rbW2     = (const float*)d_in[17];
    const float* rb_b2    = (const float*)d_in[18];
    const float* W_final  = (const float*)d_in[19];
    const float* b_final  = (const float*)d_in[20];
    const float* ra0W1    = (const float*)d_in[21];
    const float* ra0b1    = (const float*)d_in[22];
    const float* ra0W2    = (const float*)d_in[23];
    const float* ra0b2    = (const float*)d_in[24];
    const float* ra1W1    = (const float*)d_in[25];
    const float* ra1b1    = (const float*)d_in[26];
    const float* ra1W2    = (const float*)d_in[27];
    const float* ra1b2    = (const float*)d_in[28];

    // Workspace layout
    char* ws = (char*)d_ws;
    size_t off = 0;
    auto take = [&](size_t bytes) -> char* {
        char* p = ws + off;
        off += (bytes + 255) & ~(size_t)255;
        return p;
    };
    bf16_t* WT_kj    = (bf16_t*)take(EMBED * EMBED * 2);
    bf16_t* WT_down  = (bf16_t*)take(EMBED * ANGLE * 2);
    bf16_t* WT_up    = (bf16_t*)take(ANGLE * EMBED * 2);
    bf16_t* WT_ji    = (bf16_t*)take(EMBED * EMBED * 2);
    bf16_t* WT_rb1   = (bf16_t*)take(EMBED * EMBED * 2);
    bf16_t* WT_rb2   = (bf16_t*)take(EMBED * EMBED * 2);
    bf16_t* WT_fin   = (bf16_t*)take(EMBED * EMBED * 2);
    bf16_t* WT_ra0w1 = (bf16_t*)take(EMBED * EMBED * 2);
    bf16_t* WT_ra0w2 = (bf16_t*)take(EMBED * EMBED * 2);
    bf16_t* WT_ra1w1 = (bf16_t*)take(EMBED * EMBED * 2);
    bf16_t* WT_ra1w2 = (bf16_t*)take(EMBED * EMBED * 2);
    float*  Wc_rbf   = (float*) take(RBF_DIM * EMBED * 4);

    bf16_t* bufA = (bf16_t*)take((size_t)E_EDGES * EMBED * 2);
    bf16_t* bufB = (bf16_t*)take((size_t)E_EDGES * EMBED * 2);
    bf16_t* bufC = (bf16_t*)take((size_t)E_EDGES * EMBED * 2);   // m_input bf16
    bf16_t* bufM = (bf16_t*)take((size_t)E_EDGES * EMBED * 2);
    bf16_t* bufD = (bf16_t*)take((size_t)E_EDGES * ANGLE * 2);   // m_down
    double* agg  = (double*)take((size_t)E_EDGES * ANGLE * 8);

    const dim3 blk(256);
    const dim3 gGemm(E_EDGES / 128);
    const int tGrid = (128 * 128 + 255) / 256;

    // ---- weight prep (runs every call; deterministic) ----
    transpose_to_bf16_k<<<tGrid, blk, 0, stream>>>(W_kj,    WT_kj,    EMBED, EMBED);
    transpose_to_bf16_k<<<(EMBED*ANGLE+255)/256, blk, 0, stream>>>(W_down, WT_down, EMBED, ANGLE);
    transpose_to_bf16_k<<<(ANGLE*EMBED+255)/256, blk, 0, stream>>>(W_up,   WT_up,   ANGLE, EMBED);
    transpose_to_bf16_k<<<tGrid, blk, 0, stream>>>(W_ji,    WT_ji,    EMBED, EMBED);
    transpose_to_bf16_k<<<tGrid, blk, 0, stream>>>(rbW1,    WT_rb1,   EMBED, EMBED);
    transpose_to_bf16_k<<<tGrid, blk, 0, stream>>>(rbW2,    WT_rb2,   EMBED, EMBED);
    transpose_to_bf16_k<<<tGrid, blk, 0, stream>>>(W_final, WT_fin,   EMBED, EMBED);
    transpose_to_bf16_k<<<tGrid, blk, 0, stream>>>(ra0W1,   WT_ra0w1, EMBED, EMBED);
    transpose_to_bf16_k<<<tGrid, blk, 0, stream>>>(ra0W2,   WT_ra0w2, EMBED, EMBED);
    transpose_to_bf16_k<<<tGrid, blk, 0, stream>>>(ra1W1,   WT_ra1w1, EMBED, EMBED);
    transpose_to_bf16_k<<<tGrid, blk, 0, stream>>>(ra1W2,   WT_ra1w2, EMBED, EMBED);
    rbf_combine_k<<<(RBF_DIM*EMBED+255)/256, blk, 0, stream>>>(W_rbf1, W_rbf2, Wc_rbf);

    // ---- edge-level pipeline ----
    f32_to_bf16_k<<<4096, blk, 0, stream>>>(m_input, bufC, (long long)E_EDGES * EMBED);
    rbf_expand_k<<<E_EDGES / 2, blk, 0, stream>>>(rbf, Wc_rbf, bufB);

    // H = bf16( swish(m_in @ W_kj + b_kj) * rbf_e )
    gemm_bf16_k<128, 128, true, 2, false><<<gGemm, blk, 0, stream>>>(bufC, WT_kj, b_kj, bufB, bufA);
    // m_down = bf16( swish(H @ W_down) )
    gemm_bf16_k<128, 64, false, 1, false><<<gGemm, blk, 0, stream>>>(bufA, WT_down, nullptr, nullptr, bufD);

    // ---- triplet stage with f64 scatter-sum ----
    zero_f64_k<<<4096, blk, 0, stream>>>(agg, (long long)E_EDGES * ANGLE);
    triplet_k<<<T_TRIP / 256, blk, 0, stream>>>(sbf, red_ji, exp_kj, bufD, W_sbf1, W_sbf2, agg);
    f64_to_bf16_k<<<4096, blk, 0, stream>>>(agg, bufA, (long long)E_EDGES * ANGLE);

    // prop = bf16( swish(agg @ W_up) )
    gemm_bf16_k<64, 128, false, 1, false><<<gGemm, blk, 0, stream>>>(bufA, WT_up, nullptr, nullptr, bufB);
    // m = bf16( swish(m_in @ W_ji + b_ji) + prop )
    gemm_bf16_k<128, 128, true, 3, false><<<gGemm, blk, 0, stream>>>(bufC, WT_ji, b_ji, bufB, bufM);

    // res_before[0]
    gemm_bf16_k<128, 128, true, 1, false><<<gGemm, blk, 0, stream>>>(bufM, WT_rb1, rb_b1, nullptr, bufA);
    gemm_bf16_k<128, 128, true, 3, false><<<gGemm, blk, 0, stream>>>(bufA, WT_rb2, rb_b2, bufM, bufB);

    // m = bf16( swish(m @ W_final + b_final) + m_input )
    gemm_bf16_k<128, 128, true, 3, false><<<gGemm, blk, 0, stream>>>(bufB, WT_fin, b_final, bufC, bufM);

    // res_after[0]
    gemm_bf16_k<128, 128, true, 1, false><<<gGemm, blk, 0, stream>>>(bufM, WT_ra0w1, ra0b1, nullptr, bufA);
    gemm_bf16_k<128, 128, true, 3, false><<<gGemm, blk, 0, stream>>>(bufA, WT_ra0w2, ra0b2, bufM, bufB);

    // res_after[1]  (final output in f32 to d_out)
    gemm_bf16_k<128, 128, true, 1, false><<<gGemm, blk, 0, stream>>>(bufB, WT_ra1w1, ra1b1, nullptr, bufA);
    gemm_bf16_k<128, 128, true, 3, true ><<<gGemm, blk, 0, stream>>>(bufA, WT_ra1w2, ra1b2, bufB, d_out);
}
